// bPC_SNN_1563368096089
// MI455X (gfx1250) — compile-verified, matrix operand order checked by reading, not run on hardware
//
#include <hip/hip_runtime.h>

// ---------------------------------------------------------------------------
// bPC-SNN forward, MI455X (gfx1250): bf16 WMMA GEMMs, double-buffered LDS
// pipeline with async global->LDS copies (ASYNCcnt) + fused LIF/error kernels
// ---------------------------------------------------------------------------

typedef __attribute__((ext_vector_type(16))) __bf16        v16bf;
typedef __attribute__((ext_vector_type(8)))  float         v8f;
typedef __attribute__((ext_vector_type(4)))  unsigned int  uint4v;

#define BATCH 128
#define DD0   2048
#define DD1   4096
#define DD2   4096
#define DD3   512
#define NSTEPS 16

// GEMM tiling: block = 256 threads = 8 waves (wave32).
// Block tile M=128 (8 wave-stripes of 16 rows), N=64, K-step 32, 2-deep LDS.
#define BM 128
#define BN 64
#define BK 32
#define SA 48   // LDS row stride (halves) for A tile: 32 + 16 pad (bank spread)
#define SB 48   // LDS row stride (halves) for B tile (stored N-major)

#if __has_builtin(__builtin_amdgcn_global_load_async_to_lds_b128) && \
    __has_builtin(__builtin_amdgcn_s_wait_asynccnt)
#define HAVE_ASYNC_LDS 1
#else
#define HAVE_ASYNC_LDS 0
#endif

#if HAVE_ASYNC_LDS
// Builtin signature (probe-verified): param0 is a non-const pointer to
// GCC-vector 'int __vector(4)' in the global (__device__) address space.
typedef int gcc_v4i __attribute__((vector_size(16)));
typedef __attribute__((address_space(1))) gcc_v4i g_v4i;   // global
typedef __attribute__((address_space(3))) gcc_v4i l_v4i;   // LDS
// 16B global -> LDS, no VGPR staging; tracked by ASYNCcnt.
__device__ __forceinline__ void async_cp16(unsigned short* lds_p,
                                           const unsigned short* g) {
  // Int-to-pointer casts are legal into any address space; flat LDS pointers
  // carry the DS byte offset in their low 32 bits.
  __builtin_amdgcn_global_load_async_to_lds_b128(
      (g_v4i*)(unsigned long long)g,
      (l_v4i*)(unsigned int)(unsigned long long)lds_p, 0, 0);
}
#endif

__device__ __forceinline__ unsigned short f2bf(float f) {
  unsigned int u = __builtin_bit_cast(unsigned int, f);
  u += 0x7FFFu + ((u >> 16) & 1u);          // round-to-nearest-even
  return (unsigned short)(u >> 16);
}

__device__ __forceinline__ v16bf ld_frag16(const unsigned short* p) {
  union { uint4v q[2]; v16bf v; } u;
  u.q[0] = *(const uint4v*)(p);             // ds_load_b128
  u.q[1] = *(const uint4v*)(p + 8);         // ds_load_b128
  return u.v;
}

// C[128 x N] (+)= A[128 x K](bf16) * B(bf16)
//   TRANSB==0: B stored row-major K x N  (weight used "forward")
//   TRANSB==1: B stored row-major N x K  (weight used transposed, A @ W^T)
//   ACC   ==1: accumulate into existing C
template <int TRANSB, int ACC>
__global__ __launch_bounds__(256, 2) void gemm_bf16_wmma(
    const unsigned short* __restrict__ A,
    const unsigned short* __restrict__ Bm,
    float* __restrict__ C, int K, int N) {
  __shared__ __align__(16) unsigned short As[2][BM * SA];
  __shared__ __align__(16) unsigned short Bs[2][BN * SB];

  const int t    = threadIdx.x;
  const int wave = t >> 5;
  const int lane = t & 31;
  const int n0   = blockIdx.x * BN;
  const int lrow = lane & 15;
  const int lhi  = lane >> 4;
  const int rb   = wave * 16 + (lhi << 3);   // C row base for this lane's stripe

  // per-thread copy roles
  const int a_r  = t >> 1, a_c  = (t & 1) * 16;  // A: 32B/thread
  const int bt_n = t >> 2, bt_c = (t & 3) * 8;   // B^T: 16B/thread contiguous
  const int bn_k = t >> 3, bn_n = (t & 7) * 8;   // B:   16B/thread, LDS scatter

  v8f acc[4];
  if (ACC) {
#pragma unroll
    for (int j = 0; j < 4; ++j) {
      const int col = n0 + j * 16 + lrow;
#pragma unroll
      for (int r = 0; r < 8; ++r) acc[j][r] = C[(size_t)(rb + r) * N + col];
    }
  } else {
#pragma unroll
    for (int j = 0; j < 4; ++j)
#pragma unroll
      for (int r = 0; r < 8; ++r) acc[j][r] = 0.0f;
  }

  // 4 x v_wmma_f32_16x16x32_bf16 on one LDS buffer
  auto compute = [&](int b) {
    const v16bf afrag = ld_frag16(&As[b][(wave * 16 + lrow) * SA + lhi * 16]);
#pragma unroll
    for (int j = 0; j < 4; ++j) {
      const v16bf bfrag = ld_frag16(&Bs[b][(j * 16 + lrow) * SB + lhi * 16]);
      acc[j] = __builtin_amdgcn_wmma_f32_16x16x32_bf16(
          false, afrag, false, bfrag, (short)0, acc[j], false, false);
    }
  };

  // ---- stage tile 0 into buffer 0 ----
  {
    const unsigned short* ga = A + (size_t)a_r * K + a_c;
#if HAVE_ASYNC_LDS
    async_cp16(&As[0][a_r * SA + a_c], ga);
    async_cp16(&As[0][a_r * SA + a_c + 8], ga + 8);
#else
    const uint4v* g = (const uint4v*)ga;
    *(uint4v*)(&As[0][a_r * SA + a_c])     = g[0];
    *(uint4v*)(&As[0][a_r * SA + a_c + 8]) = g[1];
#endif
    if (TRANSB) {
      const unsigned short* gb = Bm + (size_t)(n0 + bt_n) * K + bt_c;
#if HAVE_ASYNC_LDS
      async_cp16(&Bs[0][bt_n * SB + bt_c], gb);
#else
      *(uint4v*)(&Bs[0][bt_n * SB + bt_c]) = *(const uint4v*)gb;
#endif
    } else {
      uint4v gv = *(const uint4v*)(Bm + (size_t)bn_k * N + n0 + bn_n);
      const unsigned short* gs = (const unsigned short*)&gv;
#pragma unroll
      for (int i = 0; i < 8; ++i) Bs[0][(bn_n + i) * SB + bn_k] = gs[i];
    }
  }

  // ---- steady state: unconditional next-tile fetch (last iter peeled) ----
  int buf = 0;
  int kk  = 0;
  uint4v breg, areg0, areg1;                // staging for sync / scatter paths
  for (; kk + BK < K; kk += BK, buf ^= 1) {
#if HAVE_ASYNC_LDS
    __builtin_amdgcn_s_wait_asynccnt(0);    // own async copies into LDS done
#endif
    __syncthreads();                        // everyone's tile visible

    const int kn = kk + BK;
    {                                       // fetch next tile into other buffer
      const unsigned short* ga = A + (size_t)a_r * K + kn + a_c;
      __builtin_prefetch(ga + BK, 0, 1);    // global_prefetch_b8 (speculative)
#if HAVE_ASYNC_LDS
      async_cp16(&As[buf ^ 1][a_r * SA + a_c], ga);
      async_cp16(&As[buf ^ 1][a_r * SA + a_c + 8], ga + 8);
      if (TRANSB) {
        const unsigned short* gb = Bm + (size_t)(n0 + bt_n) * K + kn + bt_c;
        async_cp16(&Bs[buf ^ 1][bt_n * SB + bt_c], gb);
        __builtin_prefetch(gb + BK, 0, 1);
      } else {
        breg = *(const uint4v*)(Bm + (size_t)(kn + bn_k) * N + n0 + bn_n);
      }
#else
      const uint4v* g = (const uint4v*)ga;
      areg0 = g[0];
      areg1 = g[1];
      if (TRANSB) {
        const unsigned short* gb = Bm + (size_t)(n0 + bt_n) * K + kn + bt_c;
        breg = *(const uint4v*)gb;
        __builtin_prefetch(gb + BK, 0, 1);
      } else {
        breg = *(const uint4v*)(Bm + (size_t)(kn + bn_k) * N + n0 + bn_n);
      }
#endif
    }

    compute(buf);

    // drain staged registers into the next buffer (overlaps compute)
#if HAVE_ASYNC_LDS
    if (!TRANSB) {
      const unsigned short* gs = (const unsigned short*)&breg;
#pragma unroll
      for (int i = 0; i < 8; ++i) Bs[buf ^ 1][(bn_n + i) * SB + bn_k] = gs[i];
    }
#else
    *(uint4v*)(&As[buf ^ 1][a_r * SA + a_c])     = areg0;
    *(uint4v*)(&As[buf ^ 1][a_r * SA + a_c + 8]) = areg1;
    if (TRANSB) {
      *(uint4v*)(&Bs[buf ^ 1][bt_n * SB + bt_c]) = breg;
    } else {
      const unsigned short* gs = (const unsigned short*)&breg;
#pragma unroll
      for (int i = 0; i < 8; ++i) Bs[buf ^ 1][(bn_n + i) * SB + bn_k] = gs[i];
    }
#endif
  }

  // ---- epilogue: last tile, no fetch ----
#if HAVE_ASYNC_LDS
  __builtin_amdgcn_s_wait_asynccnt(0);
#endif
  __syncthreads();
  compute(buf);

#pragma unroll
  for (int j = 0; j < 4; ++j) {
    const int col = n0 + j * 16 + lrow;
#pragma unroll
    for (int r = 0; r < 8; ++r) C[(size_t)(rb + r) * N + col] = acc[j][r];
  }
}

// ------------------------------ elementwise --------------------------------

__global__ void cvt_f32_bf16(const float* __restrict__ in,
                             unsigned short* __restrict__ out, int n) {
  for (int i = blockIdx.x * blockDim.x + threadIdx.x; i < n;
       i += gridDim.x * blockDim.x)
    out[i] = f2bf(in[i]);
}

// ti1 = -(eg1 + ed1); ti2 = -(eg2 + ed2)     (d1 == d2 so one pass)
__global__ void ti_init_kernel(const float* __restrict__ eg1,
                               const float* __restrict__ ed1, float* ti1,
                               const float* __restrict__ eg2,
                               const float* __restrict__ ed2, float* ti2, int n) {
  for (int i = blockIdx.x * blockDim.x + threadIdx.x; i < n;
       i += gridDim.x * blockDim.x) {
    ti1[i] = -(eg1[i] + ed1[i]);
    ti2[i] = -(eg2[i] + ed2[i]);
  }
}

// LIF update (dt=1, tau_j=10, kappa=1, tau_m=20, gamma=1, R=1, thr=1, tau_tr=20, T_r=2)
__global__ void lif_kernel(const float* __restrict__ ti, float* v, float* tr,
                           float* jc, float* rf, float* s,
                           unsigned short* __restrict__ sb, int n) {
  for (int i = blockIdx.x * blockDim.x + threadIdx.x; i < n;
       i += gridDim.x * blockDim.x) {
    float jj = jc[i];
    jj += 0.1f * (ti[i] - jj);                  // dt/tau_j * (-kappa*j + ti)
    float rr = rf[i];
    const bool refr = rr > 0.0f;
    float vv = v[i];
    const float vn = vv + 0.05f * (jj - vv);    // dt/tau_m * (-gamma*v + R*j)
    vv = refr ? vv : vn;
    const float sp = (!refr && vv > 1.0f) ? 1.0f : 0.0f;
    vv *= (1.0f - sp);
    rr = fmaxf(rr - 1.0f, 0.0f);
    if (sp > 0.0f) rr = 2.0f;
    float tt = tr[i];
    tt = tt - 0.05f * tt + sp;                  // tr - tr/tau_tr + spikes
    jc[i] = jj; v[i] = vv; rf[i] = rr; s[i] = sp; tr[i] = tt;
    sb[i] = f2bf(sp);
  }
}

// z' = z - z/tau + inp ; e = alpha*(z' - P) ; eb = bf16(e)
__global__ void zerr_kernel(float* z, const float* __restrict__ inp,
                            const float* __restrict__ P, float* e,
                            unsigned short* __restrict__ eb, float alpha,
                            float itau, int n) {
  for (int i = blockIdx.x * blockDim.x + threadIdx.x; i < n;
       i += gridDim.x * blockDim.x) {
    float zz = z[i];
    zz = zz - zz * itau + inp[i];
    z[i] = zz;
    const float ee = alpha * (zz - P[i]);
    e[i] = ee;
    eb[i] = f2bf(ee);
  }
}

// e = alpha*(a - P) ; optional bf16 copy
__global__ void aerr_kernel(const float* __restrict__ a,
                            const float* __restrict__ P, float* e,
                            unsigned short* __restrict__ eb, float alpha, int n) {
  for (int i = blockIdx.x * blockDim.x + threadIdx.x; i < n;
       i += gridDim.x * blockDim.x) {
    const float ee = alpha * (a[i] - P[i]);
    e[i] = ee;
    if (eb) eb[i] = f2bf(ee);
  }
}

// ------------------------------- host side ---------------------------------

static inline int ew_blocks(int n) {
  int b = (n + 2047) / 2048;       // ~8 elements / thread, grid-stride
  return b < 1 ? 1 : b;
}

extern "C" void kernel_launch(void* const* d_in, const int* in_sizes, int n_in,
                              void* d_out, int out_size, void* d_ws,
                              size_t ws_size, hipStream_t stream) {
  (void)in_sizes; (void)n_in; (void)out_size;
  const float* x  = (const float*)d_in[0];   // 128 x 2048
  const float* y  = (const float*)d_in[1];   // 128 x 512
  const float* W0 = (const float*)d_in[2];   // 2048 x 4096
  const float* W1 = (const float*)d_in[3];   // 4096 x 4096
  const float* W2 = (const float*)d_in[4];   // 4096 x 512
  const float* V0 = (const float*)d_in[5];   // 4096 x 2048
  const float* V1 = (const float*)d_in[6];   // 4096 x 4096
  const float* V2 = (const float*)d_in[7];   // 512  x 4096
  // d_in[8] = num_steps (device scalar, unreadable under graph capture) -> 16

  char*  base = (char*)d_ws;
  size_t off  = 0;
  auto alloc_f = [&](size_t elems) -> float* {
    float* r = (float*)(base + off);
    off += ((elems * 4 + 255) & ~(size_t)255);
    return r;
  };
  auto alloc_h = [&](size_t elems) -> unsigned short* {
    unsigned short* r = (unsigned short*)(base + off);
    off += ((elems * 2 + 255) & ~(size_t)255);
    return r;
  };

  const size_t n1 = (size_t)BATCH * DD1, n2 = (size_t)BATCH * DD2;
  const size_t n0 = (size_t)BATCH * DD0, n3 = (size_t)BATCH * DD3;

  // ---- zero-initialized section (state carried across steps) ----
  float *v1 = alloc_f(n1), *tr1 = alloc_f(n1), *j1 = alloc_f(n1),
        *r1 = alloc_f(n1), *s1 = alloc_f(n1);
  float *v2 = alloc_f(n2), *tr2 = alloc_f(n2), *j2 = alloc_f(n2),
        *r2 = alloc_f(n2), *s2 = alloc_f(n2);
  float *eg0 = alloc_f(n0), *eg1 = alloc_f(n1), *eg2 = alloc_f(n2);
  float *ed1 = alloc_f(n1), *ed2 = alloc_f(n2), *ed3 = alloc_f(n3);
  float *zd  = alloc_f(n0), *zl  = alloc_f(n3);
  unsigned short *eg0b = alloc_h(n0), *eg1b = alloc_h(n1), *ed2b = alloc_h(n2),
                 *ed3b = alloc_h(n3), *s1b = alloc_h(n1), *s2b = alloc_h(n2);
  const size_t zbytes = off;

  // ---- scratch (fully overwritten each step before use) ----
  float *ti1 = alloc_f(n1), *ti2 = alloc_f(n2);
  float *Pa = alloc_f(n0), *Pb = alloc_f(n1), *Pc = alloc_f(n2), *Pd = alloc_f(n3);
  float *cx = alloc_f(n1), *cy = alloc_f(n2);  // loop-invariant x@V0^T, y@W2^T
  unsigned short *xb = alloc_h(n0), *yb = alloc_h(n3);
  unsigned short *W0b = alloc_h((size_t)DD0 * DD1);
  unsigned short *W1b = alloc_h((size_t)DD1 * DD2);
  unsigned short *W2b = alloc_h((size_t)DD2 * DD3);
  unsigned short *V0b = alloc_h((size_t)DD1 * DD0);
  unsigned short *V1b = alloc_h((size_t)DD2 * DD1);
  unsigned short *V2b = alloc_h((size_t)DD3 * DD2);

  if (off > ws_size) return;  // workspace too small: deterministic no-op

  (void)hipMemsetAsync(base, 0, zbytes, stream);

  // one-time per launch: weights/inputs -> bf16 (208 MB fp32 read; ~9 us @23.3 TB/s)
  cvt_f32_bf16<<<ew_blocks((int)((size_t)DD0 * DD1)), 256, 0, stream>>>(W0, W0b, DD0 * DD1);
  cvt_f32_bf16<<<ew_blocks((int)((size_t)DD1 * DD2)), 256, 0, stream>>>(W1, W1b, DD1 * DD2);
  cvt_f32_bf16<<<ew_blocks((int)((size_t)DD2 * DD3)), 256, 0, stream>>>(W2, W2b, DD2 * DD3);
  cvt_f32_bf16<<<ew_blocks((int)((size_t)DD1 * DD0)), 256, 0, stream>>>(V0, V0b, DD1 * DD0);
  cvt_f32_bf16<<<ew_blocks((int)((size_t)DD2 * DD1)), 256, 0, stream>>>(V1, V1b, DD2 * DD1);
  cvt_f32_bf16<<<ew_blocks((int)((size_t)DD3 * DD2)), 256, 0, stream>>>(V2, V2b, DD3 * DD2);
  cvt_f32_bf16<<<ew_blocks((int)n0), 256, 0, stream>>>(x, xb, (int)n0);
  cvt_f32_bf16<<<ew_blocks((int)n3), 256, 0, stream>>>(y, yb, (int)n3);

  // loop-invariant GEMMs: cx = x @ V0^T (K=d0,N=d1), cy = y @ W2^T (K=d3,N=d2)
  gemm_bf16_wmma<1, 0><<<dim3(DD1 / BN), 256, 0, stream>>>(xb, V0b, cx, DD0, DD1);
  gemm_bf16_wmma<1, 0><<<dim3(DD2 / BN), 256, 0, stream>>>(yb, W2b, cy, DD3, DD2);

  for (int step = 0; step < NSTEPS; ++step) {
    // phase 1: ti = -eg - ed + (err @ weight) + (err @ weight)
    ti_init_kernel<<<ew_blocks((int)n1), 256, 0, stream>>>(eg1, ed1, ti1, eg2, ed2, ti2, (int)n1);
    gemm_bf16_wmma<0, 1><<<dim3(DD1 / BN), 256, 0, stream>>>(ed2b, V1b, ti1, DD2, DD1);
    gemm_bf16_wmma<0, 1><<<dim3(DD1 / BN), 256, 0, stream>>>(eg0b, W0b, ti1, DD0, DD1);
    gemm_bf16_wmma<0, 1><<<dim3(DD2 / BN), 256, 0, stream>>>(ed3b, V2b, ti2, DD3, DD2);
    gemm_bf16_wmma<0, 1><<<dim3(DD2 / BN), 256, 0, stream>>>(eg1b, W1b, ti2, DD1, DD2);

    // LIF updates (also emit bf16 spikes for phase-2 GEMMs)
    lif_kernel<<<ew_blocks((int)n1), 256, 0, stream>>>(ti1, v1, tr1, j1, r1, s1, s1b, (int)n1);
    lif_kernel<<<ew_blocks((int)n2), 256, 0, stream>>>(ti2, v2, tr2, j2, r2, s2, s2b, (int)n2);

    // phase 2 GEMMs (transposed-weight use)
    gemm_bf16_wmma<1, 0><<<dim3(DD0 / BN), 256, 0, stream>>>(s1b, W0b, Pa, DD1, DD0);
    gemm_bf16_wmma<1, 0><<<dim3(DD1 / BN), 256, 0, stream>>>(s2b, W1b, Pb, DD2, DD1);
    gemm_bf16_wmma<1, 0><<<dim3(DD2 / BN), 256, 0, stream>>>(s1b, V1b, Pc, DD1, DD2);
    gemm_bf16_wmma<1, 0><<<dim3(DD3 / BN), 256, 0, stream>>>(s2b, V2b, Pd, DD2, DD3);

    // error recomputation
    zerr_kernel<<<ew_blocks((int)n0), 256, 0, stream>>>(zd, x, Pa, eg0, eg0b, 0.5f, 0.1f, (int)n0);
    aerr_kernel<<<ew_blocks((int)n1), 256, 0, stream>>>(tr1, cx, ed1, (unsigned short*)nullptr, 0.5f, (int)n1);
    aerr_kernel<<<ew_blocks((int)n1), 256, 0, stream>>>(tr1, Pb, eg1, eg1b, 0.5f, (int)n1);
    aerr_kernel<<<ew_blocks((int)n2), 256, 0, stream>>>(tr2, Pc, ed2, ed2b, 0.5f, (int)n2);
    aerr_kernel<<<ew_blocks((int)n2), 256, 0, stream>>>(tr2, cy, eg2, (unsigned short*)nullptr, 0.5f, (int)n2);
    zerr_kernel<<<ew_blocks((int)n3), 256, 0, stream>>>(zl, y, Pd, ed3, ed3b, 0.5f, 0.1f, (int)n3);
  }

  (void)hipMemcpyAsync(d_out, ed3, n3 * sizeof(float), hipMemcpyDeviceToDevice, stream);
}
